// Attention_rel_31370441130192
// MI455X (gfx1250) — matrix operand order repaired and checked by reading, not compiled, not run on hardware
//
#include <hip/hip_runtime.h>

#define B_ 2
#define N_ 1024
#define C_ 512
#define H_ 8
#define D_ 64
#define SCALE_ 0.125f

typedef __attribute__((ext_vector_type(16))) __bf16 v16bf;
typedef __attribute__((ext_vector_type(8)))  float  v8f;
typedef __attribute__((ext_vector_type(4)))  int    v4i;

union Frag { v16bf v; unsigned u[8]; };

__device__ __forceinline__ unsigned pk2(float lo, float hi) {
  union { __bf16 h[2]; unsigned u; } t;
  t.h[0] = (__bf16)lo; t.h[1] = (__bf16)hi;
  return t.u;
}

__device__ __forceinline__ v8f wmma_bf16(v16bf a, v16bf b, v8f c) {
  // D = A(16x32 bf16) * B(32x16 bf16) + C(16x16 f32)
  return __builtin_amdgcn_wmma_f32_16x16x32_bf16(false, a, false, b, (short)0, c, false, false);
}

// K-index base for 16-bit A operand (ISA 7.12.2: 16x32, grouped pattern)
__device__ __forceinline__ int kA(int r, int hf) { return ((r >> 2) << 4) + (hf << 3) + ((r & 3) << 1); }
// K-index base for 16-bit B operand (32x16, same grouped pattern as A)
__device__ __forceinline__ int kB(int r, int hf) { return ((r >> 2) << 4) + (hf << 3) + ((r & 3) << 1); }

// Async global->LDS B128 (CDNA5 GLOBAL_LOAD_ASYNC_TO_LDS_B128, ASYNCcnt-tracked)
typedef __attribute__((address_space(1))) v4i GAS_v4i;
typedef __attribute__((address_space(3))) v4i LAS_v4i;
__device__ __forceinline__ void async_ld_b128(const float* g, float* l) {
  __builtin_amdgcn_global_load_async_to_lds_b128((GAS_v4i*)g, (LAS_v4i*)l, 0, 0);
}
__device__ __forceinline__ void wait_async0() {
  asm volatile("s_wait_asynccnt 0x0" ::: "memory");
}

// ---------------------------------------------------------------------------
// Kernel 1: QKV projections -> bf16, head-major. V stored transposed (d-major)
// so PV-GEMM B-fragments pack (j, j+1) pairs from contiguous memory.
// ---------------------------------------------------------------------------
__global__ void __launch_bounds__(128)
proj_qkv(const float* __restrict__ q, const float* __restrict__ k, const float* __restrict__ v,
         const float* __restrict__ Wq, const float* __restrict__ Wk, const float* __restrict__ Wv,
         const float* __restrict__ qb, const float* __restrict__ vb,
         __bf16* __restrict__ qh, __bf16* __restrict__ kh, __bf16* __restrict__ vhT)
{
  const int b  = blockIdx.y;
  const int rt = blockIdx.x >> 3;   // 16 row tiles of 64 (n)
  const int ct = blockIdx.x & 7;    // 8 col tiles of 64  (== head index)
  const float* X; const float* W; const float* bias; float scale; __bf16* out; int transp;
  if (blockIdx.z == 0)      { X = q; W = Wq; bias = qb;      scale = SCALE_; out = qh;  transp = 0; }
  else if (blockIdx.z == 1) { X = k; W = Wk; bias = nullptr; scale = 1.0f;   out = kh;  transp = 0; }
  else                      { X = v; W = Wv; bias = vb;      scale = 1.0f;   out = vhT; transp = 1; }

  // stride 36 floats = 144B (16B aligned rows for async B128; 16 distinct banks)
  __shared__ float sA[64][36];
  __shared__ float sB[64][36];

  const int tid  = threadIdx.x;
  const int lane = tid & 31;
  const int wv   = tid >> 5;
  const int m    = lane & 15;
  const int hf   = lane >> 4;
  const int n0   = rt * 64;
  const int o0   = ct * 64;

  v8f acc[4];
  for (int c = 0; c < 4; ++c) for (int t = 0; t < 8; ++t) acc[c][t] = 0.0f;

  for (int k0 = 0; k0 < C_; k0 += 32) {
    __syncthreads();
    for (int idx = tid; idx < 64 * 8; idx += 128) {
      int row = idx >> 3, c4 = (idx & 7) << 2;
      async_ld_b128(&X[(size_t)(b * N_ + n0 + row) * C_ + k0 + c4], &sA[row][c4]);
      async_ld_b128(&W[(size_t)(o0 + row) * C_ + k0 + c4],          &sB[row][c4]);
    }
    wait_async0();
    __syncthreads();
    Frag a;
#pragma unroll
    for (int r = 0; r < 8; ++r) {
      int kk = kA(r, hf);
      a.u[r] = pk2(sA[wv*16 + m][kk], sA[wv*16 + m][kk+1]);
    }
#pragma unroll
    for (int cb = 0; cb < 4; ++cb) {
      Frag bb;
#pragma unroll
      for (int r = 0; r < 8; ++r) {
        int kk = kB(r, hf);
        bb.u[r] = pk2(sB[cb*16 + m][kk], sB[cb*16 + m][kk+1]);
      }
      acc[cb] = wmma_bf16(a.v, bb.v, acc[cb]);
    }
  }

  const int h = ct;
#pragma unroll
  for (int cb = 0; cb < 4; ++cb)
#pragma unroll
    for (int vi = 0; vi < 8; ++vi) {
      int n = n0 + wv*16 + hf*8 + vi;
      int d = cb*16 + m;
      float val = acc[cb][vi];
      if (bias) val += bias[o0 + cb*16 + m];
      val *= scale;
      size_t idx = transp ? ((size_t)(b*H_ + h) * D_ + d) * N_ + n
                          : ((size_t)(b*H_ + h) * N_ + n) * D_ + d;
      out[idx] = (__bf16)val;
    }
}

// ---------------------------------------------------------------------------
// Kernel 2a: attn[b,h,i,j] = qh . kh + padding-mask bias   (batched GEMM, K=64)
// ---------------------------------------------------------------------------
__global__ void __launch_bounds__(128)
qk_scores(const __bf16* __restrict__ qh, const __bf16* __restrict__ kh,
          const float* __restrict__ kpm, float* __restrict__ attn)
{
  const int h = blockIdx.y, b = blockIdx.z;
  const int it = blockIdx.x >> 4, jt = blockIdx.x & 15;
  const int i0 = it * 64, j0 = jt * 64;
  const int tid = threadIdx.x, lane = tid & 31, wv = tid >> 5;
  const int m = lane & 15, hf = lane >> 4;

  const unsigned* q32 = (const unsigned*)qh;   // 2 bf16 per u32
  const unsigned* k32 = (const unsigned*)kh;
  const size_t head = (size_t)(b*H_ + h) * N_;

  v8f acc[4];
  for (int c = 0; c < 4; ++c) for (int t = 0; t < 8; ++t) acc[c][t] = 0.0f;

  const int i = i0 + wv*16 + m;
#pragma unroll
  for (int s = 0; s < 2; ++s) {
    Frag a;
#pragma unroll
    for (int r = 0; r < 8; ++r) {
      int kk = s*32 + kA(r, hf);
      a.u[r] = q32[(head + i) * 32 + (kk >> 1)];
    }
#pragma unroll
    for (int cb = 0; cb < 4; ++cb) {
      Frag bb;
      const int j = j0 + cb*16 + m;
#pragma unroll
      for (int r = 0; r < 8; ++r) {
        int kk = s*32 + kB(r, hf);
        bb.u[r] = k32[(head + j) * 32 + (kk >> 1)];
      }
      acc[cb] = wmma_bf16(a.v, bb.v, acc[cb]);
    }
  }

#pragma unroll
  for (int cb = 0; cb < 4; ++cb) {
    int j = j0 + cb*16 + m;
    float mj = kpm[b*N_ + j];
#pragma unroll
    for (int vi = 0; vi < 8; ++vi) {
      int ii = i0 + wv*16 + hf*8 + vi;
      float mi = kpm[b*N_ + ii];
      float pmin = fminf(mi, mj), pmax = fmaxf(mi, mj);
      float mb = (pmax < 0.0f) ? 0.0f : pmin;
      attn[(head + ii) * N_ + j] = acc[cb][vi] + mb;
    }
  }
}

// ---------------------------------------------------------------------------
// Kernel 2b: attn[b,h,i,j] += qh[b,h,i,:] . rpb[b,i,j,:]
// Head-batched: A = all 8 heads' q-row for this i (rows 8..15 zero),
// so the 512MB rpb tensor is streamed exactly ONCE and shared across heads.
// ---------------------------------------------------------------------------
__global__ void __launch_bounds__(256)
rpb_scores(const __bf16* __restrict__ qh, const float* __restrict__ rpb,
           float* __restrict__ attn)
{
  const int i = blockIdx.x, b = blockIdx.y;
  const int tid = threadIdx.x, lane = tid & 31, wv = tid >> 5;
  const int m = lane & 15, hf = lane >> 4;

  const unsigned* q32 = (const unsigned*)qh;
  Frag a[2];
#pragma unroll
  for (int s = 0; s < 2; ++s)
#pragma unroll
    for (int r = 0; r < 8; ++r) {
      int kk = s*32 + kA(r, hf);
      a[s].u[r] = (m < H_) ? q32[((size_t)(b*H_ + m) * N_ + i) * 32 + (kk >> 1)] : 0u;
    }

  const size_t rbase = (size_t)(b*N_ + i) * N_;  // rpb row base (rows of D_)
  for (int jt = wv; jt < 64; jt += 8) {
    const int j = jt*16 + m;
    if (jt + 8 < 64)
      __builtin_prefetch(&rpb[(rbase + (jt + 8)*16 + m) * D_], 0, 1);
    v8f acc; for (int t = 0; t < 8; ++t) acc[t] = 0.0f;
#pragma unroll
    for (int s = 0; s < 2; ++s) {
      Frag bb;
#pragma unroll
      for (int r = 0; r < 8; ++r) {
        int kk = s*32 + kB(r, hf);
        const float2 p = *(const float2*)&rpb[(rbase + j) * D_ + kk];
        bb.u[r] = pk2(p.x, p.y);
      }
      acc = wmma_bf16(a[s].v, bb.v, acc);
    }
    if (hf == 0) {  // valid rows: M = vi = head
#pragma unroll
      for (int vi = 0; vi < 8; ++vi)
        attn[((size_t)(b*H_ + vi) * N_ + i) * N_ + j] += acc[vi];
    }
  }
}

// ---------------------------------------------------------------------------
// Kernel 3: row softmax over j (in place; attn region of d_out is the output)
// ---------------------------------------------------------------------------
__global__ void __launch_bounds__(256)
softmax_rows(float* __restrict__ attn)
{
  float* p = attn + (size_t)blockIdx.x * N_;
  const int t = threadIdx.x;
  __shared__ float red[256];
  float4 v = ((float4*)p)[t];
  float mx = fmaxf(fmaxf(v.x, v.y), fmaxf(v.z, v.w));
  red[t] = mx; __syncthreads();
  for (int s = 128; s > 0; s >>= 1) { if (t < s) red[t] = fmaxf(red[t], red[t+s]); __syncthreads(); }
  mx = red[0]; __syncthreads();
  v.x = __expf(v.x - mx); v.y = __expf(v.y - mx);
  v.z = __expf(v.z - mx); v.w = __expf(v.w - mx);
  red[t] = v.x + v.y + v.z + v.w; __syncthreads();
  for (int s = 128; s > 0; s >>= 1) { if (t < s) red[t] += red[t+s]; __syncthreads(); }
  float inv = 1.0f / red[0];
  v.x *= inv; v.y *= inv; v.z *= inv; v.w *= inv;
  ((float4*)p)[t] = v;
}

// ---------------------------------------------------------------------------
// Kernel 4a: xpre[b,i,h*64+d] = sum_j attn[b,h,i,j] * vh[b,h,j,d]  (K=1024)
// ---------------------------------------------------------------------------
__global__ void __launch_bounds__(128)
pv_gemm(const float* __restrict__ attn, const __bf16* __restrict__ vhT,
        float* __restrict__ xpre)
{
  const int it = blockIdx.x, h = blockIdx.y, b = blockIdx.z;
  const int i0 = it * 64;
  const int tid = threadIdx.x, lane = tid & 31, wv = tid >> 5;
  const int m = lane & 15, hf = lane >> 4;

  const unsigned* v32 = (const unsigned*)vhT;
  const size_t arow  = (size_t)(b*H_ + h) * N_;
  const size_t vbase = (size_t)(b*H_ + h) * D_;

  v8f acc[4];
  for (int c = 0; c < 4; ++c) for (int t = 0; t < 8; ++t) acc[c][t] = 0.0f;

  const int i = i0 + wv*16 + m;
  for (int k0 = 0; k0 < N_; k0 += 32) {
    Frag a;
#pragma unroll
    for (int r = 0; r < 8; ++r) {
      int kk = k0 + kA(r, hf);
      const float2 p = *(const float2*)&attn[(arow + i) * N_ + kk];
      a.u[r] = pk2(p.x, p.y);
    }
#pragma unroll
    for (int cb = 0; cb < 4; ++cb) {
      Frag bb;
      int d = cb*16 + m;
#pragma unroll
      for (int r = 0; r < 8; ++r) {
        int kk = k0 + kB(r, hf);
        bb.u[r] = v32[((vbase + d) * N_ + kk) >> 1];
      }
      acc[cb] = wmma_bf16(a.v, bb.v, acc[cb]);
    }
  }

#pragma unroll
  for (int cb = 0; cb < 4; ++cb)
#pragma unroll
    for (int vi = 0; vi < 8; ++vi) {
      int ii = i0 + wv*16 + hf*8 + vi;
      int d  = cb*16 + m;
      xpre[(size_t)(b*N_ + ii) * C_ + h*D_ + d] = acc[cb][vi];
    }
}

// ---------------------------------------------------------------------------
// Kernel 4b: xpre[b,i,h*64+d] += sum_j attn[b,h,i,j] * rpb[b,i,j,d]
// Head-batched again: second (and last) full stream of rpb.
// ---------------------------------------------------------------------------
__global__ void __launch_bounds__(128)
rpb_out(const float* __restrict__ attn, const float* __restrict__ rpb,
        float* __restrict__ xpre)
{
  const int i = blockIdx.x, b = blockIdx.y;
  const int tid = threadIdx.x, lane = tid & 31, wv = tid >> 5;
  const int m = lane & 15, hf = lane >> 4;

  const size_t rbase = (size_t)(b*N_ + i) * N_;
  const int d = wv*16 + m;

  v8f acc; for (int t = 0; t < 8; ++t) acc[t] = 0.0f;

  for (int k0 = 0; k0 < N_; k0 += 32) {
    if (k0 + 32 < N_)
      __builtin_prefetch(&rpb[(rbase + k0 + 32) * D_ + d], 0, 1);
    Frag a;
#pragma unroll
    for (int r = 0; r < 8; ++r) {
      int kk = k0 + kA(r, hf);
      if (m < H_) {
        const float2 p = *(const float2*)&attn[((size_t)(b*H_ + m) * N_ + i) * N_ + kk];
        a.u[r] = pk2(p.x, p.y);
      } else a.u[r] = 0u;
    }
    Frag bb;
#pragma unroll
    for (int r = 0; r < 8; ++r) {
      int kk = k0 + kB(r, hf);
      float lo = rpb[(rbase + kk    ) * D_ + d];
      float hi = rpb[(rbase + kk + 1) * D_ + d];
      bb.u[r] = pk2(lo, hi);
    }
    acc = wmma_bf16(a.v, bb.v, acc);
  }

  if (hf == 0) {  // valid rows: M = vi = head
#pragma unroll
    for (int vi = 0; vi < 8; ++vi)
      xpre[(size_t)(b*N_ + i) * C_ + vi*D_ + d] += acc[vi];
  }
}

// ---------------------------------------------------------------------------
// Kernel 5: out = xpre @ Wo.T + bo
// ---------------------------------------------------------------------------
__global__ void __launch_bounds__(128)
out_proj(const float* __restrict__ xpre, const float* __restrict__ Wo,
         const float* __restrict__ bo, float* __restrict__ out)
{
  const int b = blockIdx.y;
  const int rt = blockIdx.x >> 3, ct = blockIdx.x & 7;
  const int n0 = rt*64, o0 = ct*64;
  __shared__ float sA[64][36];
  __shared__ float sB[64][36];
  const int tid = threadIdx.x, lane = tid & 31, wv = tid >> 5;
  const int m = lane & 15, hf = lane >> 4;

  v8f acc[4];
  for (int c = 0; c < 4; ++c) for (int t = 0; t < 8; ++t) acc[c][t] = 0.0f;

  for (int k0 = 0; k0 < C_; k0 += 32) {
    __syncthreads();
    for (int idx = tid; idx < 64 * 8; idx += 128) {
      int row = idx >> 3, c4 = (idx & 7) << 2;
      async_ld_b128(&xpre[(size_t)(b*N_ + n0 + row) * C_ + k0 + c4], &sA[row][c4]);
      async_ld_b128(&Wo[(size_t)(o0 + row) * C_ + k0 + c4],          &sB[row][c4]);
    }
    wait_async0();
    __syncthreads();
    Frag a;
#pragma unroll
    for (int r = 0; r < 8; ++r) {
      int kk = kA(r, hf);
      a.u[r] = pk2(sA[wv*16 + m][kk], sA[wv*16 + m][kk+1]);
    }
#pragma unroll
    for (int cb = 0; cb < 4; ++cb) {
      Frag bb;
#pragma unroll
      for (int r = 0; r < 8; ++r) {
        int kk = kB(r, hf);
        bb.u[r] = pk2(sB[cb*16 + m][kk], sB[cb*16 + m][kk+1]);
      }
      acc[cb] = wmma_bf16(a.v, bb.v, acc[cb]);
    }
  }

#pragma unroll
  for (int cb = 0; cb < 4; ++cb)
#pragma unroll
    for (int vi = 0; vi < 8; ++vi) {
      int n = n0 + wv*16 + hf*8 + vi;
      int o = o0 + cb*16 + m;
      out[(size_t)(b*N_ + n) * C_ + o] = acc[cb][vi] + bo[o];
    }
}

// ---------------------------------------------------------------------------
extern "C" void kernel_launch(void* const* d_in, const int* in_sizes, int n_in,
                              void* d_out, int out_size, void* d_ws, size_t ws_size,
                              hipStream_t stream) {
  (void)in_sizes; (void)n_in; (void)out_size; (void)ws_size;
  const float* q   = (const float*)d_in[0];
  const float* k   = (const float*)d_in[1];
  const float* v   = (const float*)d_in[2];
  const float* rpb = (const float*)d_in[3];
  const float* kpm = (const float*)d_in[4];
  const float* Wq  = (const float*)d_in[5];
  const float* Wk  = (const float*)d_in[6];
  const float* Wv  = (const float*)d_in[7];
  const float* qb  = (const float*)d_in[8];
  const float* vb  = (const float*)d_in[9];
  const float* Wo  = (const float*)d_in[10];
  const float* bo  = (const float*)d_in[11];

  float* x    = (float*)d_out;                 // (B,N,C)
  float* attn = x + (size_t)B_ * N_ * C_;      // (B,H,N,N), post-softmax

  __bf16* qh   = (__bf16*)d_ws;                           // 2 MB
  __bf16* kh   = qh + (size_t)B_ * H_ * N_ * D_;          // 2 MB
  __bf16* vhT  = kh + (size_t)B_ * H_ * N_ * D_;          // 2 MB (d-major)
  float*  xpre = (float*)(vhT + (size_t)B_ * H_ * N_ * D_); // 4 MB

  proj_qkv    <<<dim3(128, B_, 3),  128, 0, stream>>>(q, k, v, Wq, Wk, Wv, qb, vb, qh, kh, vhT);
  qk_scores   <<<dim3(256, H_, B_), 128, 0, stream>>>(qh, kh, kpm, attn);
  rpb_scores  <<<dim3(N_, B_),      256, 0, stream>>>(qh, rpb, attn);
  softmax_rows<<<dim3(B_ * H_ * N_),256, 0, stream>>>(attn);
  pv_gemm     <<<dim3(16, H_, B_),  128, 0, stream>>>(attn, vhT, xpre);
  rpb_out     <<<dim3(N_, B_),      128, 0, stream>>>(attn, rpb, xpre);
  out_proj    <<<dim3(128, B_),     128, 0, stream>>>(xpre, Wo, bo, x);
}
